// TemporalAttention_77008763617829
// MI455X (gfx1250) — compile-verified, matrix operand order checked by reading, not run on hardware
//
#include <hip/hip_runtime.h>

// Fused temporal attention for MI455X (gfx1250, wave32, WMMA).
// One workgroup per batch b: x[b] (T=512 x F=128) is read from HBM exactly
// once (non-temporal, streamed), held in LDS as bf16, and all four stages
// (MLP via bf16 WMMA, score, softmax, weighted sum) run out of LDS.
// HBM roofline: 512MB / 23.3TB/s ~ 22us.

#define T_DIM   512
#define F_DIM   128
#define U_DIM   64
#define XSTRIDE 136   // padded bf16 row stride for x in LDS (bank spread)
#define WSTRIDE 72    // padded bf16 row stride for W1 in LDS

typedef __attribute__((ext_vector_type(16))) __bf16 v16bf;
typedef __attribute__((ext_vector_type(8)))  float  v8f;
typedef __attribute__((ext_vector_type(4)))  float  fvec4;

union BF16x16 { v16bf v; unsigned int u[8]; };
union F32x8   { v8f   v; float        f[8]; };

__device__ __forceinline__ unsigned short f2bf(float x) {
  unsigned int u = __float_as_uint(x);
  return (unsigned short)((u + 0x7FFFu + ((u >> 16) & 1u)) >> 16); // RNE
}
__device__ __forceinline__ float bf2f(unsigned short h) {
  return __uint_as_float(((unsigned int)h) << 16);
}
// Branch-free tanh: 1 - 2/(e^{2s}+1). Saturates correctly (inf -> 1, 0 -> -1).
__device__ __forceinline__ float fast_tanh(float s) {
  float e = __expf(2.0f * s);                       // v_exp_f32
  return 1.0f - 2.0f * __builtin_amdgcn_rcpf(e + 1.0f);
}

extern "C" __global__ __launch_bounds__(256)
void temporal_attn_fused(const float* __restrict__ x,   // [B,T,F] f32
                         const float* __restrict__ W1,  // [F,U]   f32
                         const float* __restrict__ b1,  // [U]     f32
                         const float* __restrict__ W2,  // [U]     f32 (last dim 1)
                         float* __restrict__ out)       // [B,F]   f32
{
  extern __shared__ __align__(16) unsigned char smem[];
  unsigned short* xl  = (unsigned short*)smem;            // T_DIM * XSTRIDE bf16
  unsigned short* wl  = xl + T_DIM * XSTRIDE;             // F_DIM * WSTRIDE bf16
  float*          sc  = (float*)(wl + F_DIM * WSTRIDE);   // T_DIM scores
  float*          red = sc + T_DIM;                       // 512 reduce scratch

  const int tid  = threadIdx.x;
  const int lane = tid & 31;
  const int wv   = tid >> 5;     // wave id 0..7
  const int b    = blockIdx.x;

  // ---- Stage 1: cooperative loads (coalesced, non-temporal float4) ----
  const fvec4* gx = (const fvec4*)(x + (size_t)b * (T_DIM * F_DIM));
  for (int i = tid; i < (T_DIM * F_DIM) / 4; i += 256) {
    fvec4 v = __builtin_nontemporal_load(&gx[i]);   // streamed: keep L2 clean
    int e = i << 2;
    int t = e >> 7, f = e & 127;
    uint2 pk;
    pk.x = (unsigned int)f2bf(v.x) | ((unsigned int)f2bf(v.y) << 16);
    pk.y = (unsigned int)f2bf(v.z) | ((unsigned int)f2bf(v.w) << 16);
    *(uint2*)(xl + t * XSTRIDE + f) = pk;   // (t*136 + f)*2 is 8B aligned (f%4==0)
  }
  for (int i = tid; i < F_DIM * U_DIM; i += 256) {
    int k = i >> 6, n = i & 63;
    wl[k * WSTRIDE + n] = f2bf(W1[i]);
  }
  __syncthreads();

  // ---- Stage 2: scores[t] = tanh(x@W1 + b1) . W2  via bf16 WMMA ----
  const int half = lane >> 4;   // ISA 16x32 A layout: lanes16-31 carry K+8
  const int nl   = lane & 15;

  float w2r[4], b1r[4];
  #pragma unroll
  for (int nt = 0; nt < 4; ++nt) {
    w2r[nt] = W2[nt * 16 + nl];
    b1r[nt] = b1[nt * 16 + nl];
  }

  #pragma unroll 1
  for (int it = 0; it < 4; ++it) {
    const int t0 = (wv * 4 + it) * 16;

    // A fragments: 16x32 bf16, 4 k-steps covering K=0..127
    BF16x16 A[4];
    const unsigned short* xrow = xl + (size_t)(t0 + nl) * XSTRIDE;
    #pragma unroll
    for (int ks = 0; ks < 4; ++ks) {
      #pragma unroll
      for (int v = 0; v < 8; ++v) {
        int k = ks * 32 + ((v < 4) ? 0 : 16) + (half ? 8 : 0) + 2 * (v & 3);
        A[ks].u[v] = *(const unsigned int*)(xrow + k);
      }
    }

    float ps[8];
    #pragma unroll
    for (int r = 0; r < 8; ++r) ps[r] = 0.f;

    #pragma unroll
    for (int nt = 0; nt < 4; ++nt) {
      F32x8 acc;
      #pragma unroll
      for (int r = 0; r < 8; ++r) acc.f[r] = 0.f;

      #pragma unroll
      for (int ks = 0; ks < 4; ++ks) {
        BF16x16 Bf;   // 32x16 bf16 B fragment from W1 LDS
        #pragma unroll
        for (int v = 0; v < 8; ++v) {
          int k = ks * 32 + ((v < 4) ? 0 : 16) + (half ? 8 : 0) + 2 * (v & 3);
          unsigned int lo = wl[(k    ) * WSTRIDE + nt * 16 + nl];
          unsigned int hi = wl[(k + 1) * WSTRIDE + nt * 16 + nl];
          Bf.u[v] = lo | (hi << 16);
        }
        acc.v = __builtin_amdgcn_wmma_f32_16x16x32_bf16(
            false, A[ks].v, false, Bf.v, (short)0, acc.v, false, false);
      }

      // C/D layout: VGPR r -> row m = r + 8*half, col n = nt*16 + nl
      #pragma unroll
      for (int r = 0; r < 8; ++r) {
        float h = fast_tanh(acc.f[r] + b1r[nt]);
        ps[r] = fmaf(h, w2r[nt], ps[r]);
      }
    }

    // Reduce over n (16 lanes per half; masks < 16 keep halves separate)
    #pragma unroll
    for (int r = 0; r < 8; ++r) {
      float v = ps[r];
      v += __shfl_xor(v, 8, 32);
      v += __shfl_xor(v, 4, 32);
      v += __shfl_xor(v, 2, 32);
      v += __shfl_xor(v, 1, 32);
      ps[r] = v;
    }
    if (nl == 0) {
      #pragma unroll
      for (int r = 0; r < 8; ++r) sc[t0 + half * 8 + r] = ps[r];
    }
  }
  __syncthreads();

  // ---- Stage 3: softmax over T=512 (each thread owns 2 scores) ----
  float s0 = sc[tid], s1 = sc[tid + 256];
  float mx = fmaxf(s0, s1);
  #pragma unroll
  for (int mk = 16; mk >= 1; mk >>= 1) mx = fmaxf(mx, __shfl_xor(mx, mk, 32));
  if (lane == 0) red[wv] = mx;
  __syncthreads();
  if (tid == 0) {
    float t = red[0];
    #pragma unroll
    for (int i = 1; i < 8; ++i) t = fmaxf(t, red[i]);
    red[0] = t;
  }
  __syncthreads();
  mx = red[0];
  __syncthreads();

  float e0 = __expf(s0 - mx), e1 = __expf(s1 - mx);
  float sum = e0 + e1;
  #pragma unroll
  for (int mk = 16; mk >= 1; mk >>= 1) sum += __shfl_xor(sum, mk, 32);
  if (lane == 0) red[wv] = sum;
  __syncthreads();
  if (tid == 0) {
    float t = 0.f;
    #pragma unroll
    for (int i = 0; i < 8; ++i) t += red[i];
    red[0] = t;
  }
  __syncthreads();
  const float inv = __builtin_amdgcn_rcpf(red[0]);
  sc[tid] = e0 * inv;
  sc[tid + 256] = e1 * inv;
  __syncthreads();

  // ---- Stage 4: context[f] = sum_t attn[t] * x[t,f] (bf16 LDS, 2 cols/thread) ----
  const int f0 = (tid & 63) * 2;   // even column pair
  const int c  = tid >> 6;         // T slice 0..3
  float a0 = 0.f, a1 = 0.f;
  for (int t = c; t < T_DIM; t += 4) {
    unsigned int pk = *(const unsigned int*)(xl + t * XSTRIDE + f0);
    float w = sc[t];
    a0 = fmaf(w, bf2f((unsigned short)(pk & 0xFFFFu)), a0);
    a1 = fmaf(w, bf2f((unsigned short)(pk >> 16)),     a1);
  }
  red[c * 128 + f0]     = a0;
  red[c * 128 + f0 + 1] = a1;
  __syncthreads();
  if (tid < 128)
    out[(size_t)b * F_DIM + tid] =
        (red[tid] + red[128 + tid]) + (red[256 + tid] + red[384 + tid]);
}

extern "C" void kernel_launch(void* const* d_in, const int* in_sizes, int n_in,
                              void* d_out, int out_size, void* d_ws, size_t ws_size,
                              hipStream_t stream) {
  const float* x  = (const float*)d_in[0];
  const float* W1 = (const float*)d_in[1];
  const float* b1 = (const float*)d_in[2];
  const float* W2 = (const float*)d_in[3];
  // d_in[4] = b2: constant shift before softmax -> cancels, intentionally unused.
  float* out = (float*)d_out;

  const size_t shmem = (size_t)(T_DIM * XSTRIDE + F_DIM * WSTRIDE) * 2
                     + (size_t)(T_DIM + 512) * 4;   // = 161792 bytes

  (void)hipFuncSetAttribute((const void*)temporal_attn_fused,
                            hipFuncAttributeMaxDynamicSharedMemorySize,
                            (int)shmem);

  temporal_attn_fused<<<2048, 256, shmem, stream>>>(x, W1, b1, W2, out);
}
